// SlidingWindowAttention_58591943852045
// MI455X (gfx1250) — matrix operand order repaired and checked by reading, not compiled, not run on hardware
//
#include <hip/hip_runtime.h>
#include <hip/hip_bf16.h>

typedef __attribute__((ext_vector_type(16))) _Float16 v16h;
typedef __attribute__((ext_vector_type(8)))  _Float16 v8h;
typedef __attribute__((ext_vector_type(4)))  _Float16 v4h;
typedef __attribute__((ext_vector_type(8)))  float    v8f;

#define DIMSZ   1024
#define NHEADS  16
#define HD      64
#define SEQ     2048
#define BATCH   2
#define MROWS   (BATCH * SEQ)   // 4096
#define QKSCALE 0.125f          // 1/sqrt(64)
#define NFREQ   32              // HD/2 rope frequencies

#define BM  128
#define BN  128
#define BK  32
#define LDP 40                  // padded LDS row stride (halves)

__device__ __forceinline__ v16h mk16(v8h lo, v8h hi) {
  v16h r;
#pragma unroll
  for (int i = 0; i < 8; ++i) { r[i] = lo[i]; r[i + 8] = hi[i]; }
  return r;
}

__device__ __forceinline__ float red_max16(float v) {
#pragma unroll
  for (int m = 1; m < 16; m <<= 1) v = fmaxf(v, __shfl_xor(v, m, 32));
  return v;
}
__device__ __forceinline__ float red_sum16(float v) {
#pragma unroll
  for (int m = 1; m < 16; m <<= 1) v += __shfl_xor(v, m, 32);
  return v;
}

// ---------------------------------------------------------------- f32 -> f16
__global__ __launch_bounds__(256) void k_cvt(const float* __restrict__ in,
                                             _Float16* __restrict__ out, int n) {
  int i = (blockIdx.x * 256 + threadIdx.x) * 4;
  if (i < n) {
    float4 v = *(const float4*)(in + i);
    v4h o = {(_Float16)v.x, (_Float16)v.y, (_Float16)v.z, (_Float16)v.w};
    *(v4h*)(out + i) = o;
  }
}

// -------------------------------------------- rope cos/sin table [s][i] once
__global__ __launch_bounds__(256) void k_rope_tab(float* __restrict__ ctab,
                                                  float* __restrict__ stab) {
  int idx = blockIdx.x * 256 + threadIdx.x;       // [0, SEQ*NFREQ)
  int s = idx >> 5, i = idx & (NFREQ - 1);
  float inv_freq = expf(-(float)(2 * i) * (1.0f / 64.0f) * logf(10000.0f));
  float ang = (float)s * inv_freq;
  ctab[idx] = cosf(ang);
  stab[idx] = sinf(ang);
}

// ------------------------------------------------- shared WMMA GEMM mainloop
// C[m,n] = sum_k A[m,k] * B[n,k]   (i.e. X @ W^T), A:[M,K] B:[N,K] row-major f16
__device__ __forceinline__ void gemm_mainloop(const _Float16* __restrict__ Ag,
                                              const _Float16* __restrict__ Bg,
                                              int K, int m0, int n0,
                                              _Float16* lA, _Float16* lB,
                                              v8f acc[4][2]) {
  const int tid  = threadIdx.x;
  const int lane = tid & 31;
  const int wid  = tid >> 5;
  const int wm   = wid >> 2;       // 0..1 -> 64 rows each
  const int wn   = wid & 3;        // 0..3 -> 32 cols each
  const int r16  = lane & 15;
  const int hi   = lane >> 4;
  const int lrow = tid >> 1;             // 0..127
  const int lcol = (tid & 1) * 16;       // 0 or 16 halves

  for (int k0 = 0; k0 < K; k0 += BK) {
    v8h a0 = *(const v8h*)(Ag + (size_t)(m0 + lrow) * K + k0 + lcol);
    v8h a1 = *(const v8h*)(Ag + (size_t)(m0 + lrow) * K + k0 + lcol + 8);
    v8h b0 = *(const v8h*)(Bg + (size_t)(n0 + lrow) * K + k0 + lcol);
    v8h b1 = *(const v8h*)(Bg + (size_t)(n0 + lrow) * K + k0 + lcol + 8);
    __syncthreads();
    *(v8h*)(lA + lrow * LDP + lcol)     = a0;
    *(v8h*)(lA + lrow * LDP + lcol + 8) = a1;
    *(v8h*)(lB + lrow * LDP + lcol)     = b0;
    *(v8h*)(lB + lrow * LDP + lcol + 8) = b1;
    __syncthreads();

    v16h af[4], bf[2];
#pragma unroll
    for (int i = 0; i < 4; ++i) {
      // A-frag: row = lane&15, halves {8*hi..8*hi+7} and {16+8*hi..}
      const _Float16* p = lA + (wm * 64 + i * 16 + r16) * LDP + hi * 8;
      af[i] = mk16(*(const v8h*)p, *(const v8h*)(p + 16));
    }
#pragma unroll
    for (int jn = 0; jn < 2; ++jn) {
      // B-frag: col = lane&15, halves {8*hi..8*hi+7} and {16+8*hi..}
      const _Float16* p = lB + (wn * 32 + jn * 16 + r16) * LDP + hi * 8;
      bf[jn] = mk16(*(const v8h*)p, *(const v8h*)(p + 16));
    }
#pragma unroll
    for (int i = 0; i < 4; ++i)
#pragma unroll
      for (int jn = 0; jn < 2; ++jn)
        acc[i][jn] = __builtin_amdgcn_wmma_f32_16x16x32_f16(
            false, af[i], false, bf[jn], (short)0, acc[i][jn], false, false);
  }
}

// --------------------------------------------- QKV projection + bias + RoPE
__global__ __launch_bounds__(256) void k_qkv_rope(
    const _Float16* __restrict__ xh, const _Float16* __restrict__ wq,
    const float* __restrict__ bias, const float* __restrict__ ctab,
    const float* __restrict__ stab, _Float16* __restrict__ Qh,
    _Float16* __restrict__ Kh, _Float16* __restrict__ Vt) {
  __shared__ _Float16 lA[BM * LDP];
  __shared__ _Float16 lB[BN * LDP];
  v8f acc[4][2] = {};
  const int m0 = blockIdx.x * BM, n0 = blockIdx.y * BN;
  gemm_mainloop(xh, wq, DIMSZ, m0, n0, lA, lB, acc);

  const int lane = threadIdx.x & 31, wid = threadIdx.x >> 5;
  const int wm = wid >> 2, wn = wid & 3;
  const int r16 = lane & 15, hi = lane >> 4;
#pragma unroll
  for (int i = 0; i < 4; ++i)
#pragma unroll
    for (int jn = 0; jn < 2; ++jn) {
      const int gcol  = n0 + wn * 32 + jn * 16 + r16;   // 0..3071
      const int which = gcol >> 10;                     // 0=q 1=k 2=v
      const int d1    = gcol & 1023;
      const int head  = d1 >> 6;
      const int hd    = d1 & 63;
      const int fidx  = hd >> 1;
      const float bv  = bias[gcol];
#pragma unroll
      for (int j = 0; j < 8; ++j) {
        const int grow = m0 + wm * 64 + i * 16 + j + 8 * hi;
        const int b = grow >> 11, s = grow & 2047;
        float v = acc[i][jn][j] + bv;
        float partner = __shfl_xor(v, 1, 32);  // even/odd rope partner column
        const int bh = b * NHEADS + head;
        if (which == 2) {
          Vt[((size_t)bh * HD + hd) * SEQ + s] = (_Float16)v;   // V transposed
        } else {
          const float cs = ctab[s * NFREQ + fidx];
          const float sn = stab[s * NFREQ + fidx];
          float rot = (hd & 1) ? (partner * sn + v * cs) : (v * cs - partner * sn);
          if (which == 0)
            Qh[((size_t)bh * SEQ + s) * HD + hd] = (_Float16)(rot * QKSCALE);
          else
            Kh[((size_t)bh * SEQ + s) * HD + hd] = (_Float16)rot;
        }
      }
    }
}

// --------------------------- sliding-window flash attention, 1 wave = 16 q's
__global__ __launch_bounds__(256) void k_attn(
    const _Float16* __restrict__ Qh, const _Float16* __restrict__ Kh,
    const _Float16* __restrict__ Vt, _Float16* __restrict__ Oh,
    const int* __restrict__ winp) {
  __shared__ _Float16 lP[8 * 16 * LDP];   // per-wave 16x(32+pad) P scratch
  const int lane = threadIdx.x & 31, wid = threadIdx.x >> 5;
  const int r16 = lane & 15, hi = lane >> 4;
  const int W = winp[0];

  const int g  = blockIdx.x * 8 + wid;    // global q-tile id
  const int bh = g >> 7;                  // 128 tiles per (b,h)
  const int q0 = (g & 127) << 4;
  const int b  = bh >> 4, h = bh & 15;
  _Float16* lp = lP + wid * 16 * LDP;

  // Q A-frags (head-dim split 0..31 / 32..63), loaded once
  v16h aq0, aq1;
  {
    const _Float16* p = Qh + ((size_t)bh * SEQ + q0 + r16) * HD + hi * 8;
    aq0 = mk16(*(const v8h*)p, *(const v8h*)(p + 16));
    aq1 = mk16(*(const v8h*)(p + 32), *(const v8h*)(p + 48));
  }

  float mrun[8], lrun[8];
#pragma unroll
  for (int j = 0; j < 8; ++j) { mrun[j] = -1e30f; lrun[j] = 0.0f; }
  v8f acc[4] = {};

  int lo = q0 - (W - 1); if (lo < 0) lo = 0;
  const int kb0 = (lo >> 5) << 5;
  for (int kb = kb0; kb <= q0 + 15; kb += 32) {
    // ---- scores for two 16-key halves
    v8f sc[2];
#pragma unroll
    for (int u = 0; u < 2; ++u) {
      const _Float16* kp =
          Kh + ((size_t)bh * SEQ + kb + u * 16 + r16) * HD + hi * 8;
      v16h blo = mk16(*(const v8h*)kp, *(const v8h*)(kp + 16));
      v16h bhi = mk16(*(const v8h*)(kp + 32), *(const v8h*)(kp + 48));
      v8f z = {};
      z = __builtin_amdgcn_wmma_f32_16x16x32_f16(false, aq0, false, blo,
                                                 (short)0, z, false, false);
      z = __builtin_amdgcn_wmma_f32_16x16x32_f16(false, aq1, false, bhi,
                                                 (short)0, z, false, false);
      sc[u] = z;
    }
    // ---- online softmax (row = j + 8*hi, col/key = lane&15)
    float p0[8], p1[8], scl[8];
#pragma unroll
    for (int j = 0; j < 8; ++j) {
      const int q  = q0 + j + 8 * hi;
      const int k0 = kb + r16, k1 = k0 + 16;
      float s0 = ((k0 <= q) && (q - k0 < W)) ? sc[0][j] : -1e30f;
      float s1 = ((k1 <= q) && (q - k1 < W)) ? sc[1][j] : -1e30f;
      float mnew = fmaxf(red_max16(fmaxf(s0, s1)), mrun[j]);
      float f = __expf(mrun[j] - mnew);
      p0[j] = __expf(s0 - mnew);
      p1[j] = __expf(s1 - mnew);
      lrun[j] = lrun[j] * f + red_sum16(p0[j] + p1[j]);
      mrun[j] = mnew;
      scl[j] = f;
    }
#pragma unroll
    for (int t = 0; t < 4; ++t)
#pragma unroll
      for (int j = 0; j < 8; ++j) acc[t][j] *= scl[j];

    // ---- C-layout -> A-layout transpose of P through wave-private LDS
#pragma unroll
    for (int j = 0; j < 8; ++j) {
      const int m = j + 8 * hi;
      lp[m * LDP + r16]      = (_Float16)p0[j];
      lp[m * LDP + 16 + r16] = (_Float16)p1[j];
    }
    asm volatile("s_wait_dscnt 0" ::: "memory");
    v16h pf;
    {
      const _Float16* pp = lp + r16 * LDP + hi * 8;
      pf = mk16(*(const v8h*)pp, *(const v8h*)(pp + 16));
    }
    // ---- acc += P @ V  (Vt rows are contiguous in key dim)
#pragma unroll
    for (int t = 0; t < 4; ++t) {
      const _Float16* vp =
          Vt + ((size_t)bh * HD + t * 16 + r16) * SEQ + kb + hi * 8;
      v16h vf = mk16(*(const v8h*)vp, *(const v8h*)(vp + 16));
      acc[t] = __builtin_amdgcn_wmma_f32_16x16x32_f16(false, pf, false, vf,
                                                      (short)0, acc[t], false, false);
    }
  }

  // ---- normalize + store [b, s, h*64+d] as f16
#pragma unroll
  for (int j = 0; j < 8; ++j) {
    const float inv = 1.0f / lrun[j];
    const int srow = q0 + j + 8 * hi;
#pragma unroll
    for (int t = 0; t < 4; ++t)
      Oh[((size_t)(b * SEQ + srow)) * DIMSZ + h * HD + t * 16 + r16] =
          (_Float16)(acc[t][j] * inv);
  }
}

// ------------------------------------------------------- output projection
__global__ __launch_bounds__(256) void k_out_gemm(
    const _Float16* __restrict__ Ah, const _Float16* __restrict__ Wh,
    const float* __restrict__ bias, float* __restrict__ out) {
  __shared__ _Float16 lA[BM * LDP];
  __shared__ _Float16 lB[BN * LDP];
  v8f acc[4][2] = {};
  const int m0 = blockIdx.x * BM, n0 = blockIdx.y * BN;
  gemm_mainloop(Ah, Wh, DIMSZ, m0, n0, lA, lB, acc);

  const int lane = threadIdx.x & 31, wid = threadIdx.x >> 5;
  const int wm = wid >> 2, wn = wid & 3;
  const int r16 = lane & 15, hi = lane >> 4;
#pragma unroll
  for (int i = 0; i < 4; ++i)
#pragma unroll
    for (int jn = 0; jn < 2; ++jn) {
      const int gcol = n0 + wn * 32 + jn * 16 + r16;
      const float bv = bias[gcol];
#pragma unroll
      for (int j = 0; j < 8; ++j) {
        const int grow = m0 + wm * 64 + i * 16 + j + 8 * hi;
        out[(size_t)grow * DIMSZ + gcol] = acc[i][jn][j] + bv;
      }
    }
}

extern "C" void kernel_launch(void* const* d_in, const int* in_sizes, int n_in,
                              void* d_out, int out_size, void* d_ws, size_t ws_size,
                              hipStream_t stream) {
  const float* x     = (const float*)d_in[0];
  const float* qkv_w = (const float*)d_in[1];
  const float* qkv_b = (const float*)d_in[2];
  const float* out_w = (const float*)d_in[3];
  const float* out_b = (const float*)d_in[4];
  const int*   win   = (const int*)d_in[5];

  char* ws = (char*)d_ws;
  size_t off = 0;
  auto balloc = [&](size_t bytes) { char* p = ws + off; off += (bytes + 255) & ~(size_t)255; return p; };
  _Float16* xh  = (_Float16*)balloc((size_t)MROWS * DIMSZ * 2);     // x f16 (reused as Oh)
  _Float16* wqh = (_Float16*)balloc((size_t)3 * DIMSZ * DIMSZ * 2); // qkv_w f16
  _Float16* woh = (_Float16*)balloc((size_t)DIMSZ * DIMSZ * 2);     // out_w f16
  _Float16* Qh  = (_Float16*)balloc((size_t)MROWS * DIMSZ * 2);     // [b,h,s,d], pre-scaled
  _Float16* Kh  = (_Float16*)balloc((size_t)MROWS * DIMSZ * 2);     // [b,h,s,d]
  _Float16* Vt  = (_Float16*)balloc((size_t)MROWS * DIMSZ * 2);     // [b,h,d,s]
  float*    ctab = (float*)balloc((size_t)SEQ * NFREQ * 4);         // rope cos
  float*    stab = (float*)balloc((size_t)SEQ * NFREQ * 4);         // rope sin
  _Float16* Oh  = xh;                                               // attn out reuses xh

  const int nx = MROWS * DIMSZ;
  const int nw = 3 * DIMSZ * DIMSZ;
  const int no = DIMSZ * DIMSZ;
  k_cvt<<<nx / 1024, 256, 0, stream>>>(x, xh, nx);
  k_cvt<<<nw / 1024, 256, 0, stream>>>(qkv_w, wqh, nw);
  k_cvt<<<no / 1024, 256, 0, stream>>>(out_w, woh, no);
  k_rope_tab<<<(SEQ * NFREQ) / 256, 256, 0, stream>>>(ctab, stab);

  dim3 gA(MROWS / BM, (3 * DIMSZ) / BN);
  k_qkv_rope<<<gA, 256, 0, stream>>>(xh, wqh, qkv_b, ctab, stab, Qh, Kh, Vt);

  k_attn<<<(BATCH * NHEADS * (SEQ / 16)) / 8, 256, 0, stream>>>(Qh, Kh, Vt, Oh, win);

  dim3 gC(MROWS / BM, DIMSZ / BN);
  k_out_gemm<<<gC, 256, 0, stream>>>(Oh, woh, out_b, (float*)d_out);
}
